// CESTNet_23802708754485
// MI455X (gfx1250) — matrix operand order repaired and checked by reading, not compiled, not run on hardware
//
#include <hip/hip_runtime.h>

typedef _Float16 v16h __attribute__((ext_vector_type(16)));
typedef float    v8f  __attribute__((ext_vector_type(8)));

#define NB   8
#define HGT  96
#define WID  96
#define CCH  128
#define LTOK (HGT*WID)           /* 9216  */
#define ROWS (NB*LTOK)           /* 73728 */
#define WINS (NB*144)            /* 1152  */
#define QSZ  ((size_t)WINS*4*64*32)  /* 9437184 halves per q/k/v region */

#define OFF_X 0ull
#define OFF_W 37748736ull        /* ROWS*128*4                */
#define OFF_B 38797312ull        /* + 1 MiB weights           */
#define OFF_C 114294784ull       /* + ROWS*512*2              */
#define WS_NEED 133169152ull     /* + 18874368                */

static __device__ __forceinline__ v8f wmma16(v16h a, v16h b, v8f c) {
  return __builtin_amdgcn_wmma_f32_16x16x32_f16(false, a, false, b, (short)0, c, false, false);
}

/* A fragment: 16x32 f16 tile, row-major memory, contiguous-k layout:
   lanes 0-15: M=lane, K = 0..15; lanes 16-31: M=lane-16, K = 16..31 */
static __device__ __forceinline__ v16h frag_a(const _Float16* __restrict__ base, int ld,
                                              int kofs, int lane) {
  const int r  = lane & 15;
  const int kh = (lane >> 4) & 1;
  const _Float16* p = base + (size_t)r * ld + kofs + kh * 16;
  v16h f;
#pragma unroll
  for (int j = 0; j < 16; ++j) f[j] = p[j];
  return f;
}

/* B fragment from N-major memory (mem[n][k], row-major, like transposed weight):
   contiguous-k layout: lanes 0-15 hold K=0..15, lanes 16-31 hold K=16..31, N = lane&15 */
static __device__ __forceinline__ v16h frag_b_nm(const _Float16* __restrict__ base, int ld,
                                                 int kofs, int lane) {
  const int n  = lane & 15;
  const int kh = (lane >> 4) & 1;
  const _Float16* p = base + (size_t)n * ld + kofs + kh * 16;
  v16h f;
#pragma unroll
  for (int j = 0; j < 16; ++j) f[j] = p[j];
  return f;
}

/* B fragment from K-major memory (mem[k][n], row-major, e.g. V matrix) */
static __device__ __forceinline__ v16h frag_b_km(const _Float16* __restrict__ base, int ld,
                                                 int kofs, int lane) {
  const int n  = lane & 15;
  const int kh = (lane >> 4) & 1;
  const _Float16* p = base + n + (size_t)(kofs + kh * 16) * ld;
  v16h f;
#pragma unroll
  for (int j = 0; j < 16; ++j) f[j] = p[(size_t)j * ld];
  return f;
}

/* ---------------- weight convert: w[K][N] f32 -> wT[N][K] f16 ---------------- */
__global__ void k_wconv(const float* __restrict__ w, _Float16* __restrict__ wT, int K, int N) {
  const int idx = blockIdx.x * 256 + threadIdx.x;
  if (idx >= K * N) return;
  const int n = idx / K, k = idx - n * K;
  wT[idx] = (_Float16)w[(size_t)k * N + n];
}

__global__ void k_copy(const float* __restrict__ s, float* __restrict__ d, int n) {
  const int i = blockIdx.x * 256 + threadIdx.x;
  if (i < n) d[i] = s[i];
}

/* ---------------- LN1 + window gather (+roll) + QKV GEMM ---------------- */
__global__ __launch_bounds__(256) void k_ln_qkv(
    const float* __restrict__ x, const float* __restrict__ g, const float* __restrict__ bt,
    const _Float16* __restrict__ wT, const float* __restrict__ bias,
    _Float16* __restrict__ qkv, int shift) {
  __shared__ _Float16 sA[64 * 128];
  const int win = blockIdx.x;
  const int tid = threadIdx.x, lane = tid & 31, wave = tid >> 5;
  const int b = win / 144, wi = win % 144, wh = wi / 12, wwc = wi % 12;
  { /* LayerNorm: 4 lanes per row */
    const int row = tid >> 2, qd = tid & 3;
    const int r = row >> 3, c = row & 7;
    const int hh = (wh * 8 + r + shift) % HGT;
    const int wp = (wwc * 8 + c + shift) % WID;
    const float* xr = x + (((size_t)b * LTOK) + (size_t)hh * WID + wp) * CCH + qd * 32;
    const float4* x4 = (const float4*)xr;
    float vb[32], s = 0.f, s2 = 0.f;
#pragma unroll
    for (int j = 0; j < 8; ++j) {
      float4 f = x4[j];
      vb[4*j] = f.x; vb[4*j+1] = f.y; vb[4*j+2] = f.z; vb[4*j+3] = f.w;
      s  += f.x + f.y + f.z + f.w;
      s2 += f.x*f.x + f.y*f.y + f.z*f.z + f.w*f.w;
    }
    s  += __shfl_xor(s, 1);  s  += __shfl_xor(s, 2);
    s2 += __shfl_xor(s2, 1); s2 += __shfl_xor(s2, 2);
    const float m  = s * (1.f / 128.f);
    const float ri = rsqrtf(s2 * (1.f / 128.f) - m * m + 1e-5f);
#pragma unroll
    for (int j = 0; j < 32; ++j) {
      const int cc = qd * 32 + j;
      sA[row * 128 + cc] = (_Float16)((vb[j] - m) * ri * g[cc] + bt[cc]);
    }
  }
  __syncthreads();
  const int mtile = wave & 3, nr = lane & 15, mhi = lane >> 4;
  v16h afr[4];
#pragma unroll
  for (int ks = 0; ks < 4; ++ks) afr[ks] = frag_a(&sA[mtile * 16 * 128], 128, ks * 32, lane);
  for (int nt = (wave >> 2); nt < 24; nt += 2) {
    const float bv = bias[nt * 16 + nr];
    v8f acc = {bv, bv, bv, bv, bv, bv, bv, bv};
#pragma unroll
    for (int ks = 0; ks < 4; ++ks) {
      v16h bfr = frag_b_nm(wT + (size_t)(nt * 16) * 128, 128, ks * 32, lane);
      acc = wmma16(afr[ks], bfr, acc);
    }
    const int col = nt * 16 + nr;
    const int which = col >> 7, cc = col & 127, head = cc >> 5, dd = cc & 31;
    const float sc = (which == 0) ? 0.17677669529663687f : 1.0f; /* q * hd^-0.5 */
    _Float16* op = qkv + (size_t)which * QSZ + ((size_t)(win * 4 + head) * 64) * 32 + dd;
#pragma unroll
    for (int i = 0; i < 8; ++i) {
      const int tok = mtile * 16 + mhi * 8 + i;
      op[(size_t)tok * 32] = (_Float16)(acc[i] * sc);
    }
  }
}

static __device__ __forceinline__ int regid8(int wh, int wwc, int tok) {
  const int hp = wh * 8 + (tok >> 3), wp = wwc * 8 + (tok & 7);
  const int rh = (hp < HGT - 8) ? 0 : ((hp < HGT - 4) ? 1 : 2);
  const int rw = (wp < WID - 8) ? 0 : ((wp < WID - 4) ? 1 : 2);
  return rh * 3 + rw;
}

/* ---------------- per-(window,head) attention: qk^T, mask, softmax, PV ---------------- */
__global__ __launch_bounds__(32) void k_attn(const _Float16* __restrict__ qkv,
                                             _Float16* __restrict__ aout, int shift) {
  __shared__ float sS[64 * 64];
  const int wid4 = blockIdx.x;
  const int win = wid4 >> 2, head = wid4 & 3;
  const int lane = threadIdx.x, nr = lane & 15, mhi = lane >> 4;
  const _Float16* qp = qkv + ((size_t)(win * 4 + head) * 64) * 32;
  const _Float16* kp = qp + QSZ;
  const _Float16* vp = qp + 2 * QSZ;
  v16h kfr[4];
#pragma unroll
  for (int nt = 0; nt < 4; ++nt) kfr[nt] = frag_b_nm(kp + nt * 16 * 32, 32, 0, lane);
#pragma unroll
  for (int mt = 0; mt < 4; ++mt) {
    v16h qfr = frag_a(qp + mt * 16 * 32, 32, 0, lane);
#pragma unroll
    for (int nt = 0; nt < 4; ++nt) {
      v8f acc = {0.f, 0.f, 0.f, 0.f, 0.f, 0.f, 0.f, 0.f};
      acc = wmma16(qfr, kfr[nt], acc);
#pragma unroll
      for (int i = 0; i < 8; ++i)
        sS[(mt * 16 + mhi * 8 + i) * 64 + nt * 16 + nr] = acc[i];
    }
  }
  __syncthreads();
  const int wi = win % 144, wh = wi / 12, wwc = wi % 12;
  const int c0 = lane * 2, c1 = c0 + 1;
  int id0 = 0, id1 = 0;
  if (shift > 0) { id0 = regid8(wh, wwc, c0); id1 = regid8(wh, wwc, c1); }
  _Float16* sP = (_Float16*)sS;
  for (int r = 0; r < 64; ++r) {   /* softmax row r; f16 probs written in place (safe) */
    float x0 = sS[r * 64 + c0], x1 = sS[r * 64 + c1];
    if (shift > 0) {
      const int idr = regid8(wh, wwc, r);
      if (idr != id0) x0 = -3.0e38f;
      if (idr != id1) x1 = -3.0e38f;
    }
    float mx = fmaxf(x0, x1);
#pragma unroll
    for (int m = 1; m < 32; m <<= 1) mx = fmaxf(mx, __shfl_xor(mx, m));
    const float e0 = __expf(x0 - mx), e1 = __expf(x1 - mx);
    float sm = e0 + e1;
#pragma unroll
    for (int m = 1; m < 32; m <<= 1) sm += __shfl_xor(sm, m);
    const float rs = 1.f / sm;
    sP[r * 64 + c0] = (_Float16)(e0 * rs);
    sP[r * 64 + c1] = (_Float16)(e1 * rs);
  }
  __syncthreads();
  v16h vfr[2][2];
#pragma unroll
  for (int nt = 0; nt < 2; ++nt)
#pragma unroll
    for (int ks = 0; ks < 2; ++ks)
      vfr[nt][ks] = frag_b_km(vp + nt * 16, 32, ks * 32, lane);
#pragma unroll
  for (int mt = 0; mt < 4; ++mt) {
    v16h p0 = frag_a(sP + mt * 16 * 64, 64, 0, lane);
    v16h p1 = frag_a(sP + mt * 16 * 64, 64, 32, lane);
#pragma unroll
    for (int nt = 0; nt < 2; ++nt) {
      v8f acc = {0.f, 0.f, 0.f, 0.f, 0.f, 0.f, 0.f, 0.f};
      acc = wmma16(p0, vfr[nt][0], acc);
      acc = wmma16(p1, vfr[nt][1], acc);
#pragma unroll
      for (int i = 0; i < 8; ++i) {
        const int tok = mt * 16 + mhi * 8 + i;
        aout[((size_t)win * 64 + tok) * 128 + head * 32 + nt * 16 + nr] = (_Float16)acc[i];
      }
    }
  }
}

/* ---------------- proj GEMM + un-roll scatter + residual ---------------- */
__global__ __launch_bounds__(256) void k_proj(const _Float16* __restrict__ aout,
    const _Float16* __restrict__ wT, const float* __restrict__ bias,
    float* __restrict__ x, int shift) {
  const int win = blockIdx.x;
  const int tid = threadIdx.x, lane = tid & 31, wave = tid >> 5;
  const int b = win / 144, wi = win % 144, wh = wi / 12, wwc = wi % 12;
  const int mtile = wave & 3, nr = lane & 15, mhi = lane >> 4;
  const _Float16* Ab = aout + (size_t)win * 64 * 128 + (size_t)mtile * 16 * 128;
  v16h afr[4];
#pragma unroll
  for (int ks = 0; ks < 4; ++ks) afr[ks] = frag_a(Ab, 128, ks * 32, lane);
  for (int nt = wave >> 2; nt < 8; nt += 2) {
    const float bv = bias[nt * 16 + nr];
    v8f acc = {bv, bv, bv, bv, bv, bv, bv, bv};
#pragma unroll
    for (int ks = 0; ks < 4; ++ks) {
      v16h bfr = frag_b_nm(wT + (size_t)(nt * 16) * 128, 128, ks * 32, lane);
      acc = wmma16(afr[ks], bfr, acc);
    }
#pragma unroll
    for (int i = 0; i < 8; ++i) {
      const int tok = mtile * 16 + mhi * 8 + i;
      const int hh = (wh * 8 + (tok >> 3) + shift) % HGT;
      const int wp = (wwc * 8 + (tok & 7) + shift) % WID;
      const size_t idx = (((size_t)b * LTOK) + (size_t)hh * WID + wp) * CCH + nt * 16 + nr;
      x[idx] += acc[i];
    }
  }
}

/* ---------------- LN2 + fc1 + exact GELU ---------------- */
__global__ __launch_bounds__(256) void k_mlp1(
    const float* __restrict__ x, const float* __restrict__ g, const float* __restrict__ bt,
    const _Float16* __restrict__ wT, const float* __restrict__ bias,
    _Float16* __restrict__ hbuf) {
  __shared__ _Float16 sA[64 * 128];
  const size_t row0 = (size_t)blockIdx.x * 64;
  const int tid = threadIdx.x, lane = tid & 31, wave = tid >> 5;
  {
    const int row = tid >> 2, qd = tid & 3;
    const float4* x4 = (const float4*)(x + (row0 + row) * CCH + qd * 32);
    float vb[32], s = 0.f, s2 = 0.f;
#pragma unroll
    for (int j = 0; j < 8; ++j) {
      float4 f = x4[j];
      vb[4*j] = f.x; vb[4*j+1] = f.y; vb[4*j+2] = f.z; vb[4*j+3] = f.w;
      s  += f.x + f.y + f.z + f.w;
      s2 += f.x*f.x + f.y*f.y + f.z*f.z + f.w*f.w;
    }
    s  += __shfl_xor(s, 1);  s  += __shfl_xor(s, 2);
    s2 += __shfl_xor(s2, 1); s2 += __shfl_xor(s2, 2);
    const float m  = s * (1.f / 128.f);
    const float ri = rsqrtf(s2 * (1.f / 128.f) - m * m + 1e-5f);
#pragma unroll
    for (int j = 0; j < 32; ++j) {
      const int cc = qd * 32 + j;
      sA[row * 128 + cc] = (_Float16)((vb[j] - m) * ri * g[cc] + bt[cc]);
    }
  }
  __syncthreads();
  const int mtile = wave & 3, nr = lane & 15, mhi = lane >> 4;
  v16h afr[4];
#pragma unroll
  for (int ks = 0; ks < 4; ++ks) afr[ks] = frag_a(&sA[mtile * 16 * 128], 128, ks * 32, lane);
  for (int nt = wave >> 2; nt < 32; nt += 2) {
    const float bv = bias[nt * 16 + nr];
    v8f acc = {bv, bv, bv, bv, bv, bv, bv, bv};
#pragma unroll
    for (int ks = 0; ks < 4; ++ks) {
      v16h bfr = frag_b_nm(wT + (size_t)(nt * 16) * 128, 128, ks * 32, lane);
      acc = wmma16(afr[ks], bfr, acc);
    }
#pragma unroll
    for (int i = 0; i < 8; ++i) {
      const int tok = mtile * 16 + mhi * 8 + i;
      const float v = acc[i];
      const float ge = 0.5f * v * (1.f + erff(v * 0.70710678118654752f));
      hbuf[(row0 + tok) * 512 + nt * 16 + nr] = (_Float16)ge;
    }
  }
}

/* ---------------- fc2 + residual ---------------- */
__global__ __launch_bounds__(256) void k_mlp2(const _Float16* __restrict__ hbuf,
    const _Float16* __restrict__ wT, const float* __restrict__ bias, float* __restrict__ x) {
  const size_t row0 = (size_t)blockIdx.x * 64;
  const int tid = threadIdx.x, lane = tid & 31, wave = tid >> 5;
  const int mtile = wave & 3, nr = lane & 15, mhi = lane >> 4;
  const _Float16* Ab = hbuf + (row0 + (size_t)mtile * 16) * 512;
  for (int nt = wave >> 2; nt < 8; nt += 2) {
    const float bv = bias[nt * 16 + nr];
    v8f acc = {bv, bv, bv, bv, bv, bv, bv, bv};
#pragma unroll
    for (int ks = 0; ks < 16; ++ks) {
      v16h af = frag_a(Ab, 512, ks * 32, lane);
      v16h bf = frag_b_nm(wT + (size_t)(nt * 16) * 512, 512, ks * 32, lane);
      acc = wmma16(af, bf, acc);
    }
#pragma unroll
    for (int i = 0; i < 8; ++i) {
      const int tok = mtile * 16 + mhi * 8 + i;
      x[(row0 + tok) * 128 + nt * 16 + nr] += acc[i];
    }
  }
}

/* ---------------- patch-merge gather + LN ---------------- */
__global__ __launch_bounds__(256) void k_pmln(const float* __restrict__ x,
    const float* __restrict__ g, const float* __restrict__ bt, _Float16* __restrict__ xcln) {
  const int tid = threadIdx.x;
  const int row = blockIdx.x * 32 + (tid >> 3);
  const int sub = tid & 7;
  const int b = row / 2304, rr = row % 2304, i2 = rr / 48, j2 = rr % 48;
  const int chunk = sub >> 1, c0 = (sub & 1) * 64;
  const int hs = 2 * i2 + (chunk & 1), wsrc = 2 * j2 + (chunk >> 1);
  const float4* s4 = (const float4*)(x + (((size_t)b * LTOK) + (size_t)hs * WID + wsrc) * CCH + c0);
  float vb[64], s = 0.f, s2 = 0.f;
#pragma unroll
  for (int j = 0; j < 16; ++j) {
    float4 f = s4[j];
    vb[4*j] = f.x; vb[4*j+1] = f.y; vb[4*j+2] = f.z; vb[4*j+3] = f.w;
    s  += f.x + f.y + f.z + f.w;
    s2 += f.x*f.x + f.y*f.y + f.z*f.z + f.w*f.w;
  }
  s  += __shfl_xor(s, 1);  s  += __shfl_xor(s, 2);  s  += __shfl_xor(s, 4);
  s2 += __shfl_xor(s2, 1); s2 += __shfl_xor(s2, 2); s2 += __shfl_xor(s2, 4);
  const float m  = s * (1.f / 512.f);
  const float ri = rsqrtf(s2 * (1.f / 512.f) - m * m + 1e-5f);
#pragma unroll
  for (int t = 0; t < 64; ++t) {
    const int j = sub * 64 + t;
    xcln[(size_t)row * 512 + j] = (_Float16)((vb[t] - m) * ri * g[j] + bt[j]);
  }
}

/* ---------------- reduction GEMM (512 -> 256), fp32 out ---------------- */
__global__ __launch_bounds__(256) void k_pmgemm(const _Float16* __restrict__ xcln,
    const _Float16* __restrict__ wT, float* __restrict__ out) {
  const size_t row0 = (size_t)blockIdx.x * 64;
  const int tid = threadIdx.x, lane = tid & 31, wave = tid >> 5;
  const int mtile = wave & 3, nr = lane & 15, mhi = lane >> 4;
  const _Float16* Ab = xcln + (row0 + (size_t)mtile * 16) * 512;
  for (int nt = wave >> 2; nt < 16; nt += 2) {
    v8f acc = {0.f, 0.f, 0.f, 0.f, 0.f, 0.f, 0.f, 0.f};
#pragma unroll
    for (int ks = 0; ks < 16; ++ks) {
      v16h af = frag_a(Ab, 512, ks * 32, lane);
      v16h bf = frag_b_nm(wT + (size_t)(nt * 16) * 512, 512, ks * 32, lane);
      acc = wmma16(af, bf, acc);
    }
#pragma unroll
    for (int i = 0; i < 8; ++i)
      out[(row0 + mtile * 16 + mhi * 8 + i) * 256 + nt * 16 + nr] = acc[i];
  }
}

extern "C" void kernel_launch(void* const* d_in, const int* in_sizes, int n_in,
                              void* d_out, int out_size, void* d_ws, size_t ws_size,
                              hipStream_t stream) {
  (void)in_sizes; (void)n_in; (void)out_size;
  if (ws_size < WS_NEED) return;
  char* ws = (char*)d_ws;
  float*    xbuf = (float*)(ws + OFF_X);
  _Float16* wreg = (_Float16*)(ws + OFF_W);
  _Float16* bigB = (_Float16*)(ws + OFF_B);  /* qkv region OR mlp hidden */
  _Float16* bufC = (_Float16*)(ws + OFF_C);  /* attn out OR merged-LN    */

  _Float16* wqkv[2]  = { wreg + 0,      wreg + 196608 };
  _Float16* wproj[2] = { wreg + 49152,  wreg + 245760 };
  _Float16* wfc1[2]  = { wreg + 65536,  wreg + 262144 };
  _Float16* wfc2[2]  = { wreg + 131072, wreg + 327680 };
  _Float16* wred     = wreg + 393216;

  auto wc = [&](int idx, _Float16* dst, int K, int N) {
    const int n = K * N;
    k_wconv<<<(n + 255) / 256, 256, 0, stream>>>((const float*)d_in[idx], dst, K, N);
  };
  wc(3,  wqkv[0], 128, 384);  wc(5,  wproj[0], 128, 128);
  wc(9,  wfc1[0], 128, 512);  wc(11, wfc2[0], 512, 128);
  wc(15, wqkv[1], 128, 384);  wc(17, wproj[1], 128, 128);
  wc(21, wfc1[1], 128, 512);  wc(23, wfc2[1], 512, 128);
  wc(27, wred, 512, 256);

  k_copy<<<(ROWS * CCH + 255) / 256, 256, 0, stream>>>((const float*)d_in[0], xbuf, ROWS * CCH);

  for (int blk = 0; blk < 2; ++blk) {
    const int shift = blk ? 4 : 0;
    const int o = 1 + blk * 12;
    const float* ln1g = (const float*)d_in[o + 0];
    const float* ln1b = (const float*)d_in[o + 1];
    const float* qkvb = (const float*)d_in[o + 3];
    const float* projb= (const float*)d_in[o + 5];
    const float* ln2g = (const float*)d_in[o + 6];
    const float* ln2b = (const float*)d_in[o + 7];
    const float* fc1b = (const float*)d_in[o + 9];
    const float* fc2b = (const float*)d_in[o + 11];

    k_ln_qkv<<<WINS, 256, 0, stream>>>(xbuf, ln1g, ln1b, wqkv[blk], qkvb, bigB, shift);
    k_attn  <<<WINS * 4, 32, 0, stream>>>(bigB, bufC, shift);
    k_proj  <<<WINS, 256, 0, stream>>>(bufC, wproj[blk], projb, xbuf, shift);
    k_mlp1  <<<WINS, 256, 0, stream>>>(xbuf, ln2g, ln2b, wfc1[blk], fc1b, bigB);
    k_mlp2  <<<WINS, 256, 0, stream>>>(bigB, wfc2[blk], fc2b, xbuf);
  }
  k_pmln  <<<576, 256, 0, stream>>>(xbuf, (const float*)d_in[25], (const float*)d_in[26], bufC);
  k_pmgemm<<<288, 256, 0, stream>>>(bufC, wred, (float*)d_out);
}
